// GCNModelVAE_5875515261564
// MI455X (gfx1250) — compile-verified, matrix operand order checked by reading, not run on hardware
//
#include <hip/hip_runtime.h>
#include <hip/hip_bf16.h>

#define N_NODES 2048
#define F_IN    768
#define H1      512
#define H2      256
#define LC      256
#define MAX_K   512
#define CAND_CAP 8192

typedef __attribute__((ext_vector_type(16))) __bf16 v16bf;
typedef __attribute__((ext_vector_type(8)))  float  v8f;
typedef __attribute__((ext_vector_type(4)))  int    v4i;

__device__ __forceinline__ float leakyf(float x) { return x > 0.f ? x : 0.01f * x; }

// ---------------- convert f32 -> bf16 ----------------
__global__ void cvt_bf16_kernel(const float* __restrict__ x, __bf16* __restrict__ y, int n) {
  int i = blockIdx.x * blockDim.x + threadIdx.x;
  int stride = gridDim.x * blockDim.x;
  for (; i < n; i += stride) y[i] = (__bf16)x[i];
}

// ---------------- CDNA5 async global->LDS copy (16 bytes / lane) ----------------
__device__ __forceinline__ void async_copy16(const __bf16* g, __bf16* l) {
  __builtin_amdgcn_global_load_async_to_lds_b128((v4i*)(void*)g, (v4i*)(void*)l, 0, 0);
}

// ---------------- WMMA bf16 GEMM ----------------
// C[M,N] = epi(alpha * A[M,K] x op(B));  op(B)=B[K,N] if TRANSB==0 else B[N,K]
// EPI: 0 none, 1 leaky-relu, 2 sigmoid
#define BM 128
#define BN 128
#define TK 32
#define LDK (TK + 8)   // row stride 40 elems = 80B (16B multiple)

__device__ __forceinline__ v16bf load_frag(const __bf16* p) {
  // ISA 16-bit A/B layout: elem j = K[8*half + j], elem 8+j = K[16 + 8*half + j]
  v16bf f;
  #pragma unroll
  for (int j = 0; j < 8; ++j) { f[j] = p[j]; f[8 + j] = p[16 + j]; }
  return f;
}

template<int TRANSB, int EPI>
__global__ __launch_bounds__(256)
void gemm_bf16_kernel(const __bf16* __restrict__ A, int lda,
                      const __bf16* __restrict__ B, int ldb,
                      float* __restrict__ C, int ldc,
                      int K, float alpha) {
  __shared__ __bf16 sA[2][BM][LDK];
  __shared__ __bf16 sB[2][BN][LDK];   // stored [n][k]
  const int tid  = threadIdx.x;
  const int lane = tid & 31;
  const int wave = tid >> 5;
  const int wm   = wave >> 1;          // 0..3 : 32-row band
  const int wn   = wave & 1;           // 0..1 : 64-col band
  const int half = lane >> 4;
  const int l15  = lane & 15;
  const int rowBase = blockIdx.y * BM;
  const int colBase = blockIdx.x * BN;

  // staging coordinates
  const int am  = tid >> 1, ak  = (tid & 1) * 16;   // A: 128x32, 16 elems (2 async b128) / thread
  const int bn  = tid >> 1, bk  = (tid & 1) * 16;   // B transB: same pattern
  const int bkr = tid >> 3, bnr = (tid & 7) * 16;   // B normal: row k=bkr, 16 contiguous cols

  v8f acc[2][4] = {};
  v16bf breg = {};

  const int nIters = K / TK;

  // ---- prologue: stage tile 0 ----
  {
    const __bf16* asrc = A + (size_t)(rowBase + am) * lda + ak;
    async_copy16(asrc,     &sA[0][am][ak]);
    async_copy16(asrc + 8, &sA[0][am][ak + 8]);
    if (TRANSB) {
      const __bf16* bsrc = B + (size_t)(colBase + bn) * ldb + bk;
      async_copy16(bsrc,     &sB[0][bn][bk]);
      async_copy16(bsrc + 8, &sB[0][bn][bk + 8]);
    } else {
      breg = *(const v16bf*)(B + (size_t)bkr * ldb + colBase + bnr);
      #pragma unroll
      for (int j = 0; j < 16; ++j) sB[0][bnr + j][bkr] = breg[j];
    }
  }
  __builtin_amdgcn_s_wait_asynccnt(0);
  __syncthreads();

  for (int it = 0; it < nIters; ++it) {
    const int cur = it & 1, nxt = cur ^ 1;
    const bool more = (it + 1 < nIters);
    const int k1 = (it + 1) * TK;

    // issue next tile's global traffic before the math
    if (more) {
      const __bf16* asrc = A + (size_t)(rowBase + am) * lda + k1 + ak;
      async_copy16(asrc,     &sA[nxt][am][ak]);
      async_copy16(asrc + 8, &sA[nxt][am][ak + 8]);
      if (TRANSB) {
        const __bf16* bsrc = B + (size_t)(colBase + bn) * ldb + k1 + bk;
        async_copy16(bsrc,     &sB[nxt][bn][bk]);
        async_copy16(bsrc + 8, &sB[nxt][bn][bk + 8]);
      } else {
        breg = *(const v16bf*)(B + (size_t)(k1 + bkr) * ldb + colBase + bnr);
      }
    }

    // 8 WMMAs per wave per K-step
    const v16bf a0 = load_frag(&sA[cur][wm * 32 +  0 + l15][8 * half]);
    const v16bf a1 = load_frag(&sA[cur][wm * 32 + 16 + l15][8 * half]);
    #pragma unroll
    for (int s = 0; s < 4; ++s) {
      const v16bf b = load_frag(&sB[cur][wn * 64 + s * 16 + l15][8 * half]);
      acc[0][s] = __builtin_amdgcn_wmma_f32_16x16x32_bf16(false, a0, false, b, (short)0, acc[0][s], false, false);
      acc[1][s] = __builtin_amdgcn_wmma_f32_16x16x32_bf16(false, a1, false, b, (short)0, acc[1][s], false, false);
    }

    // commit transposed B registers to the next LDS buffer
    if (!TRANSB && more) {
      #pragma unroll
      for (int j = 0; j < 16; ++j) sB[nxt][bnr + j][bkr] = breg[j];
    }

    __builtin_amdgcn_s_wait_asynccnt(0);
    __syncthreads();
  }

  // C/D layout: VGPR r -> row r (lanes 0-15) / 8+r (lanes 16-31), col = l15
  #pragma unroll
  for (int t = 0; t < 2; ++t) {
    const int m0 = rowBase + wm * 32 + t * 16 + 8 * half;
    const int n0 = colBase + wn * 64 + l15;
    #pragma unroll
    for (int s = 0; s < 4; ++s) {
      #pragma unroll
      for (int r = 0; r < 8; ++r) {
        float v = acc[t][s][r] * alpha;
        if (EPI == 1) v = leakyf(v);
        else if (EPI == 2) v = 1.f / (1.f + __expf(-v));
        C[(size_t)(m0 + r) * ldc + n0 + s * 16] = v;
      }
    }
  }
}

// ---------------- softmax over 256-wide rows (wave per row) ----------------
__global__ __launch_bounds__(256)
void softmax_rows256_kernel(const float* __restrict__ scores, __bf16* __restrict__ attn, int rows) {
  const int wave = threadIdx.x >> 5, lane = threadIdx.x & 31;
  const int row  = blockIdx.x * 8 + wave;
  if (row >= rows) return;
  const float* x = scores + (size_t)row * 256;
  float v[8], mx = -1e30f;
  #pragma unroll
  for (int i = 0; i < 8; ++i) { v[i] = x[i * 32 + lane]; mx = fmaxf(mx, v[i]); }
  #pragma unroll
  for (int m = 16; m >= 1; m >>= 1) mx = fmaxf(mx, __shfl_xor(mx, m, 32));
  float s = 0.f;
  #pragma unroll
  for (int i = 0; i < 8; ++i) { v[i] = __expf(v[i] - mx); s += v[i]; }
  #pragma unroll
  for (int m = 16; m >= 1; m >>= 1) s += __shfl_xor(s, m, 32);
  const float inv = 1.f / s;
  __bf16* y = attn + (size_t)row * 256;
  #pragma unroll
  for (int i = 0; i < 8; ++i) y[i * 32 + lane] = (__bf16)(v[i] * inv);
}

// ---------------- reparameterize ----------------
__global__ void z_kernel(const float* __restrict__ mu, const float* __restrict__ lv,
                         const float* __restrict__ eps, __bf16* __restrict__ zb, int n) {
  int i = blockIdx.x * blockDim.x + threadIdx.x;
  int stride = gridDim.x * blockDim.x;
  for (; i < n; i += stride)
    zb[i] = (__bf16)(eps[i] * __expf(0.5f * lv[i]) + mu[i]);
}

// ---------------- deterministic reductions ----------------
__global__ __launch_bounds__(256)
void sum_partial_kernel(const float* __restrict__ x, long n, float* __restrict__ part) {
  long i = (long)blockIdx.x * blockDim.x + threadIdx.x;
  const long stride = (long)gridDim.x * blockDim.x;
  float s = 0.f;
  for (; i < n; i += stride) s += x[i];
  __shared__ float sm[256];
  sm[threadIdx.x] = s; __syncthreads();
  for (int o = 128; o > 0; o >>= 1) { if (threadIdx.x < o) sm[threadIdx.x] += sm[threadIdx.x + o]; __syncthreads(); }
  if (threadIdx.x == 0) part[blockIdx.x] = sm[0];
}

__global__ __launch_bounds__(256)
void bce_partial_kernel(const float* __restrict__ p, const float* __restrict__ labels,
                        const float* __restrict__ labels_sum, long n, float* __restrict__ part) {
  const float Nf = (float)N_NODES, NN = Nf * Nf;
  const float s_lbl = *labels_sum;
  const float pos_w = (NN - s_lbl + Nf) / (s_lbl - Nf + 0.01f);
  long i = (long)blockIdx.x * blockDim.x + threadIdx.x;
  const long stride = (long)gridDim.x * blockDim.x;
  float s = 0.f;
  for (; i < n; i += stride) {
    const float pi = p[i], y = labels[i];
    const float ls_pos = -log1pf(__expf(-pi));   // log_sigmoid(pi)
    const float ls_neg = -log1pf(__expf(pi));    // log_sigmoid(-pi)
    s += -(pos_w * y * ls_pos + (1.f - y) * ls_neg);
  }
  __shared__ float sm[256];
  sm[threadIdx.x] = s; __syncthreads();
  for (int o = 128; o > 0; o >>= 1) { if (threadIdx.x < o) sm[threadIdx.x] += sm[threadIdx.x + o]; __syncthreads(); }
  if (threadIdx.x == 0) part[blockIdx.x] = sm[0];
}

__global__ __launch_bounds__(256)
void kld_partial_kernel(const float* __restrict__ mu_po, const float* __restrict__ mu_pr,
                        const float* __restrict__ lv_po, const float* __restrict__ lv_pr,
                        long n, float* __restrict__ part) {
  long i = (long)blockIdx.x * blockDim.x + threadIdx.x;
  const long stride = (long)gridDim.x * blockDim.x;
  float s = 0.f;
  for (; i < n; i += stride) {
    const float d  = mu_pr[i] - mu_po[i];
    const float dl = lv_po[i] - lv_pr[i];
    s += d * d / __expf(lv_pr[i]) + __expf(dl) - 1.f - dl;
  }
  __shared__ float sm[256];
  sm[threadIdx.x] = s; __syncthreads();
  for (int o = 128; o > 0; o >>= 1) { if (threadIdx.x < o) sm[threadIdx.x] += sm[threadIdx.x + o]; __syncthreads(); }
  if (threadIdx.x == 0) part[blockIdx.x] = sm[0];
}

__global__ __launch_bounds__(256)
void final_sum_kernel(const float* __restrict__ part, int n, float* __restrict__ out) {
  float s = 0.f;
  for (int i = threadIdx.x; i < n; i += 256) s += part[i];
  __shared__ float sm[256];
  sm[threadIdx.x] = s; __syncthreads();
  for (int o = 128; o > 0; o >>= 1) { if (threadIdx.x < o) sm[threadIdx.x] += sm[threadIdx.x + o]; __syncthreads(); }
  if (threadIdx.x == 0) *out = sm[0];
}

// ---------------- top-k over triu ----------------
__global__ void init_topk_kernel(unsigned* __restrict__ hist, unsigned* __restrict__ counter) {
  int i = blockIdx.x * blockDim.x + threadIdx.x;
  if (i < 1024) hist[i] = 0u;
  if (i == 0) *counter = 0u;
}

__global__ __launch_bounds__(256)
void hist_kernel(const float* __restrict__ p, unsigned* __restrict__ hist, long n) {
  long i = (long)blockIdx.x * blockDim.x + threadIdx.x;
  const long stride = (long)gridDim.x * blockDim.x;
  for (; i < n; i += stride) {
    const int row = (int)(i >> 11), col = (int)(i & 2047);
    if (col > row) {
      int b = (int)(p[i] * 1024.f);
      b = b < 0 ? 0 : (b > 1023 ? 1023 : b);
      atomicAdd(&hist[b], 1u);
    }
  }
}

__global__ void threshold_kernel(const unsigned* __restrict__ hist, float* __restrict__ thresh) {
  unsigned c = 0; int t = 1023;
  for (; t >= 0; --t) { c += hist[t]; if (c >= MAX_K) break; }
  if (t < 0) t = 0;
  *thresh = (float)t / 1024.f;
}

__global__ __launch_bounds__(256)
void collect_kernel(const float* __restrict__ p, const float* __restrict__ thresh,
                    float* __restrict__ cval, int* __restrict__ cidx,
                    unsigned* __restrict__ counter, long n) {
  const float th = *thresh;
  long i = (long)blockIdx.x * blockDim.x + threadIdx.x;
  const long stride = (long)gridDim.x * blockDim.x;
  for (; i < n; i += stride) {
    const int row = (int)(i >> 11), col = (int)(i & 2047);
    if (col > row && p[i] >= th) {
      unsigned pos = atomicAdd(counter, 1u);
      if (pos < CAND_CAP) { cval[pos] = p[i]; cidx[pos] = (int)i; }
    }
  }
}

__global__ __launch_bounds__(256)
void select_kernel(float* __restrict__ cval, const int* __restrict__ cidx,
                   const unsigned* __restrict__ counter, int* __restrict__ topk) {
  __shared__ float sv[256];
  __shared__ int   sl[256];   // local candidate index
  __shared__ int   sg[256];   // global flat index (tie-break)
  const unsigned c = *counter;
  const int nc = (c < (unsigned)CAND_CAP) ? (int)c : CAND_CAP;
  for (int k = 0; k < MAX_K; ++k) {
    float bv = -1.f; int bl = -1, bg = 0x7fffffff;
    for (int i = threadIdx.x; i < nc; i += 256) {
      const float v = cval[i]; const int g = cidx[i];
      if (v > bv || (v == bv && g < bg)) { bv = v; bl = i; bg = g; }
    }
    sv[threadIdx.x] = bv; sl[threadIdx.x] = bl; sg[threadIdx.x] = bg;
    __syncthreads();
    for (int o = 128; o > 0; o >>= 1) {
      if (threadIdx.x < o) {
        const float v1 = sv[threadIdx.x + o];
        if (v1 > sv[threadIdx.x] || (v1 == sv[threadIdx.x] && sg[threadIdx.x + o] < sg[threadIdx.x])) {
          sv[threadIdx.x] = v1; sl[threadIdx.x] = sl[threadIdx.x + o]; sg[threadIdx.x] = sg[threadIdx.x + o];
        }
      }
      __syncthreads();
    }
    if (threadIdx.x == 0) {
      const int b = sl[0];
      topk[k] = (b >= 0) ? cidx[b] : 0;
      if (b >= 0) cval[b] = -2.f;
    }
    __syncthreads();
  }
}

__global__ void gather_kernel(const int* __restrict__ topk, const float* __restrict__ ns2,
                              float* __restrict__ out_rel) {
  const int k = blockIdx.x, j = threadIdx.x;
  const int idx = topk[k];
  const int a = idx >> 11, b = idx & 2047;
  out_rel[(size_t)k * H2 + j] = ns2[(size_t)a * H2 + j] + ns2[(size_t)b * H2 + j];
}

__global__ void finalize_kernel(const float* __restrict__ lsum, const float* __restrict__ bsum,
                                const float* __restrict__ ksum, float* __restrict__ out) {
  const int tid = threadIdx.x;
  const size_t base = (size_t)MAX_K * H2;
  if (tid < MAX_K) out[base + tid] = 0.f;   // rel_mask: all False
  if (tid == 0) {
    const float Nf = (float)N_NODES, NN = Nf * Nf;
    const float s = *lsum;
    const float norm = NN / (NN - s + Nf);
    out[base + MAX_K]     = norm * (*bsum) / NN;          // recons
    out[base + MAX_K + 1] = 0.5f / Nf * ((*ksum) / Nf);   // kld
  }
}

// ---------------- host ----------------
extern "C" void kernel_launch(void* const* d_in, const int* in_sizes, int n_in,
                              void* d_out, int out_size, void* d_ws, size_t ws_size,
                              hipStream_t stream) {
  const float* ns_emb = (const float*)d_in[0];
  const float* adj    = (const float*)d_in[1];
  const float* adjp   = (const float*)d_in[2];
  const float* cond   = (const float*)d_in[3];   // [1, LC, F_IN]
  const float* labels = (const float*)d_in[4];
  const float* eps    = (const float*)d_in[5];
  const float* W_map  = (const float*)d_in[6];
  const float* postW[7], * priorW[7];
  for (int i = 0; i < 7; ++i) { postW[i] = (const float*)d_in[7 + i]; priorW[i] = (const float*)d_in[14 + i]; }
  float* out = (float*)d_out;

  // ---- bump allocator over d_ws ----
  char* base = (char*)d_ws; size_t off = 0;
  auto alloc  = [&](size_t bytes) { off = (off + 255) & ~(size_t)255; void* p = base + off; off += bytes; return p; };
  auto allocF = [&](size_t n) { return (float*)alloc(n * 4); };
  auto allocB = [&](size_t n) { return (__bf16*)alloc(n * 2); };

  const size_t NN2 = (size_t)N_NODES * N_NODES;

  __bf16* adj_bf  = allocB(NN2);
  __bf16* adjp_bf = allocB(NN2);
  __bf16* ns_bf   = allocB((size_t)N_NODES * F_IN);
  __bf16* cond_bf = allocB((size_t)LC * F_IN);
  __bf16* Wmap_bf = allocB((size_t)F_IN * H2);
  const int wdim[7][2] = { {F_IN,H1},{H1,H1},{F_IN,H1},{F_IN,H1},{H1,H1},{H1,H2},{H1,H2} };
  __bf16* postWb[7]; __bf16* priorWb[7];
  for (int i = 0; i < 7; ++i) postWb[i]  = allocB((size_t)wdim[i][0] * wdim[i][1]);
  for (int i = 0; i < 7; ++i) priorWb[i] = allocB((size_t)wdim[i][0] * wdim[i][1]);

  float* mu_po = allocF((size_t)N_NODES * H2);
  float* lv_po = allocF((size_t)N_NODES * H2);
  float* mu_pr = allocF((size_t)N_NODES * H2);
  float* lv_pr = allocF((size_t)N_NODES * H2);

  float*  Sf    = allocF((size_t)N_NODES * H1);  __bf16* S_bf  = allocB((size_t)N_NODES * H1);
  float*  Hf    = allocF((size_t)N_NODES * H1);  __bf16* H_bf  = allocB((size_t)N_NODES * H1);
  float*  Qf    = allocF((size_t)N_NODES * H1);  __bf16* Q_bf  = allocB((size_t)N_NODES * H1);
  float*  Kf    = allocF((size_t)LC * H1);       __bf16* K_bf  = allocB((size_t)LC * H1);
  float*  Vf    = allocF((size_t)LC * H1);       __bf16* V_bf  = allocB((size_t)LC * H1);
  float*  scores  = allocF((size_t)2 * N_NODES * 256);
  __bf16* attn_bf = allocB((size_t)2 * N_NODES * 256);
  float*  Of    = allocF((size_t)N_NODES * H1);  __bf16* O_bf   = allocB((size_t)N_NODES * H1);
  float*  mhaF  = allocF((size_t)N_NODES * H1);  __bf16* mha_bf = allocB((size_t)N_NODES * H1);
  float*  MxF   = allocF((size_t)N_NODES * H2);  __bf16* Mx_bf  = allocB((size_t)N_NODES * H2);

  __bf16* z_bf    = allocB((size_t)N_NODES * H2);
  float*  recover = allocF(NN2);
  float*  ns2     = allocF((size_t)N_NODES * H2);

  float* part0 = allocF(1024);
  float* part1 = allocF(1024);
  float* part2 = allocF(1024);
  float* labels_sum = allocF(1);
  float* bce_sum    = allocF(1);
  float* kld_sum    = allocF(1);
  float* thresh     = allocF(1);
  unsigned* hist    = (unsigned*)alloc(1024 * 4);
  unsigned* counter = (unsigned*)alloc(4);
  float*    cand_val = allocF(CAND_CAP);
  int*      cand_idx = (int*)alloc(CAND_CAP * 4);
  int*      topk     = (int*)alloc(MAX_K * 4);

  auto cvt = [&](const float* x, __bf16* y, size_t n) {
    long blocks = ((long)n + 255) / 256; if (blocks > 4096) blocks = 4096;
    cvt_bf16_kernel<<<(int)blocks, 256, 0, stream>>>(x, y, (int)n);
  };
  auto gemm = [&](const __bf16* A, int lda, const __bf16* B, int ldb, float* C, int ldc,
                  int M, int Nn, int K, int transB, int ep, float alpha) {
    dim3 g(Nn / BN, M / BM);
    if (!transB && ep == 0)
      gemm_bf16_kernel<0, 0><<<g, 256, 0, stream>>>(A, lda, B, ldb, C, ldc, K, alpha);
    else if (!transB && ep == 1)
      gemm_bf16_kernel<0, 1><<<g, 256, 0, stream>>>(A, lda, B, ldb, C, ldc, K, alpha);
    else if (transB && ep == 0)
      gemm_bf16_kernel<1, 0><<<g, 256, 0, stream>>>(A, lda, B, ldb, C, ldc, K, alpha);
    else
      gemm_bf16_kernel<1, 2><<<g, 256, 0, stream>>>(A, lda, B, ldb, C, ldc, K, alpha);
  };

  // ---- upfront conversions ----
  init_topk_kernel<<<4, 256, 0, stream>>>(hist, counter);
  cvt(adj,    adj_bf,  NN2);
  cvt(adjp,   adjp_bf, NN2);
  cvt(ns_emb, ns_bf,  (size_t)N_NODES * F_IN);
  cvt(cond,   cond_bf,(size_t)LC * F_IN);
  cvt(W_map,  Wmap_bf,(size_t)F_IN * H2);
  for (int i = 0; i < 7; ++i) cvt(postW[i],  postWb[i],  (size_t)wdim[i][0] * wdim[i][1]);
  for (int i = 0; i < 7; ++i) cvt(priorW[i], priorWb[i], (size_t)wdim[i][0] * wdim[i][1]);

  // ---- encoder ----
  auto run_encoder = [&](const __bf16* adj_b, __bf16* const* Wb, float* mu_out, float* lv_out) {
    // gcn(ns_emb, W_hid, adj)
    gemm(ns_bf, F_IN, Wb[0], H1, Sf, H1, N_NODES, H1, F_IN, 0, 1, 1.f);
    cvt(Sf, S_bf, (size_t)N_NODES * H1);
    gemm(adj_b, N_NODES, S_bf, H1, Hf, H1, N_NODES, H1, N_NODES, 0, 1, 1.f);
    cvt(Hf, H_bf, (size_t)N_NODES * H1);
    // mha
    gemm(H_bf, H1, Wb[1], H1, Qf, H1, N_NODES, H1, H1, 0, 0, 1.f);
    cvt(Qf, Q_bf, (size_t)N_NODES * H1);
    gemm(cond_bf, F_IN, Wb[2], H1, Kf, H1, LC, H1, F_IN, 0, 0, 1.f);
    cvt(Kf, K_bf, (size_t)LC * H1);
    gemm(cond_bf, F_IN, Wb[3], H1, Vf, H1, LC, H1, F_IN, 0, 0, 1.f);
    cvt(Vf, V_bf, (size_t)LC * H1);
    for (int h = 0; h < 2; ++h)   // scores_h = Q_h K_h^T / 16
      gemm(Q_bf + h * 256, H1, K_bf + h * 256, H1, scores + (size_t)h * N_NODES * 256, 256,
           N_NODES, 256, 256, 1, 0, 0.0625f);
    softmax_rows256_kernel<<<(2 * N_NODES) / 8, 256, 0, stream>>>(scores, attn_bf, 2 * N_NODES);
    for (int h = 0; h < 2; ++h)   // O_h = attn_h V_h
      gemm(attn_bf + (size_t)h * N_NODES * 256, 256, V_bf + h * 256, H1, Of + h * 256, H1,
           N_NODES, 256, LC, 0, 0, 1.f);
    cvt(Of, O_bf, (size_t)N_NODES * H1);
    gemm(O_bf, H1, Wb[4], H1, mhaF, H1, N_NODES, H1, H1, 0, 0, 1.f);
    cvt(mhaF, mha_bf, (size_t)N_NODES * H1);
    // gcn(hidden, W_mu, adj)
    gemm(mha_bf, H1, Wb[5], H2, MxF, H2, N_NODES, H2, H1, 0, 1, 1.f);
    cvt(MxF, Mx_bf, (size_t)N_NODES * H2);
    gemm(adj_b, N_NODES, Mx_bf, H2, mu_out, H2, N_NODES, H2, N_NODES, 0, 1, 1.f);
    // gcn(hidden, W_var, adj)
    gemm(mha_bf, H1, Wb[6], H2, MxF, H2, N_NODES, H2, H1, 0, 1, 1.f);
    cvt(MxF, Mx_bf, (size_t)N_NODES * H2);
    gemm(adj_b, N_NODES, Mx_bf, H2, lv_out, H2, N_NODES, H2, N_NODES, 0, 1, 1.f);
  };

  run_encoder(adj_bf,  postWb,  mu_po, lv_po);
  run_encoder(adjp_bf, priorWb, mu_pr, lv_pr);

  // ---- decode ----
  z_kernel<<<1024, 256, 0, stream>>>(mu_po, lv_po, eps, z_bf, N_NODES * H2);
  gemm(z_bf, H2, z_bf, H2, recover, N_NODES, N_NODES, N_NODES, H2, 1, 2, 1.f); // sigmoid(z z^T)
  gemm(ns_bf, F_IN, Wmap_bf, H2, ns2, H2, N_NODES, H2, F_IN, 0, 1, 1.f);      // leaky(ns @ W_map)

  // ---- losses (deterministic two-pass reductions) ----
  sum_partial_kernel<<<1024, 256, 0, stream>>>(labels, (long)NN2, part0);
  final_sum_kernel<<<1, 256, 0, stream>>>(part0, 1024, labels_sum);
  bce_partial_kernel<<<1024, 256, 0, stream>>>(recover, labels, labels_sum, (long)NN2, part1);
  final_sum_kernel<<<1, 256, 0, stream>>>(part1, 1024, bce_sum);
  kld_partial_kernel<<<1024, 256, 0, stream>>>(mu_po, mu_pr, lv_po, lv_pr, (long)N_NODES * H2, part2);
  final_sum_kernel<<<1, 256, 0, stream>>>(part2, 1024, kld_sum);

  // ---- top-512 over triu(recover) ----
  hist_kernel<<<2048, 256, 0, stream>>>(recover, hist, (long)NN2);
  threshold_kernel<<<1, 1, 0, stream>>>(hist, thresh);
  collect_kernel<<<2048, 256, 0, stream>>>(recover, thresh, cand_val, cand_idx, counter, (long)NN2);
  select_kernel<<<1, 256, 0, stream>>>(cand_val, cand_idx, counter, topk);
  gather_kernel<<<MAX_K, 256, 0, stream>>>(topk, ns2, out);

  finalize_kernel<<<1, 512, 0, stream>>>(labels_sum, bce_sum, kld_sum, out);
}
